// GATLayer_36696200577051
// MI455X (gfx1250) — matrix operand order repaired and checked by reading, not compiled, not run on hardware
//
#include <hip/hip_runtime.h>
#include <hip/hip_bf16.h>
#include <math.h>

// GAT layer for MI455X (gfx1250, wave32).
//  K1: h = x @ W via v_wmma_f32_16x16x32_bf16 (W staged in LDS as bf16)
//  K2: per-node attention logits alpha_s/alpha_d
//  K3: init out=bias, max=-inf, denom=0
//  K4: segment-max over edges (global_atomic_max_num_f32)
//  K5: exp + segment-sum (global_atomic_add_f32), store per-edge exp
//  K6: normalized weighted scatter-add into out

typedef __attribute__((ext_vector_type(16))) __bf16 v16bf;
typedef __attribute__((ext_vector_type(8)))  float  v8f;

#define GAT_N      50000
#define GAT_E      800000
#define GAT_K      128
#define GAT_OC     64          // HEADS * OUT_CH
#define GAT_EP     (GAT_E + GAT_N)
#define GAT_TILES  (GAT_N / 16)   // 3125, exact
#define NEG_SLOPE  0.2f

// ---------------------------------------------------------------- K1: GEMM
__global__ __launch_bounds__(128) void gat_gemm_wmma(
    const float* __restrict__ x,      // [N,128]
    const float* __restrict__ W,      // [128,64]
    float* __restrict__ hf)           // [N,64]
{
    __shared__ __bf16 sW[GAT_K * GAT_OC];   // 16 KB, bf16 copy of W

    // cooperative load + convert of W into LDS
    for (int idx = threadIdx.x; idx < GAT_K * GAT_OC; idx += blockDim.x)
        sW[idx] = (__bf16)W[idx];
    __syncthreads();

    const int lane = threadIdx.x & 31;
    const int nIdx = lane & 15;            // column within 16-wide tile
    const int kHiB = (lane >> 4) << 4;     // +16 K offset for upper half-wave; unused (B uses kOffA)
    const int kOffA = (lane >> 4) << 3;    // A: +8  K offset for upper half-wave

    // Build all 16 B fragments (4 k-steps x 4 n-tiles) once per wave.
    // 16-bit B 32x16 layout: half-slot i -> K = k0 + kOffA + (i&7) + 16*(i>>3), N = nIdx.
    v16bf bfrag[4][4];
#pragma unroll
    for (int s = 0; s < 4; ++s) {
        const int k0 = s * 32;
#pragma unroll
        for (int t = 0; t < 4; ++t) {
            const int n0 = t * 16;
#pragma unroll
            for (int i = 0; i < 16; ++i)
                bfrag[s][t][i] = sW[(k0 + kOffA + (i & 7) + ((i >> 3) << 4)) * GAT_OC + n0 + nIdx];
        }
    }

    const int wid    = blockIdx.x * (blockDim.x >> 5) + (threadIdx.x >> 5);
    const int nWaves = gridDim.x * (blockDim.x >> 5);

    for (int tile = wid; tile < GAT_TILES; tile += nWaves) {   // wave-uniform
        const int m0 = tile * 16;
        const int r  = m0 + nIdx;          // A row for this lane

        v8f acc[4] = {};

#pragma unroll
        for (int s = 0; s < 4; ++s) {
            const int k0 = s * 32;
            // 16-bit A 16x32 layout: halves 0..7 -> K=k0+kOffA+i,
            //                        halves 8..15 -> K=k0+16+kOffA+i
            const float* ap = x + (size_t)r * GAT_K + k0 + kOffA;
            float4 p0 = *(const float4*)(ap);
            float4 p1 = *(const float4*)(ap + 4);
            float4 q0 = *(const float4*)(ap + 16);
            float4 q1 = *(const float4*)(ap + 20);
            v16bf a;
            a[0]=(__bf16)p0.x; a[1]=(__bf16)p0.y; a[2]=(__bf16)p0.z; a[3]=(__bf16)p0.w;
            a[4]=(__bf16)p1.x; a[5]=(__bf16)p1.y; a[6]=(__bf16)p1.z; a[7]=(__bf16)p1.w;
            a[8]=(__bf16)q0.x; a[9]=(__bf16)q0.y; a[10]=(__bf16)q0.z; a[11]=(__bf16)q0.w;
            a[12]=(__bf16)q1.x; a[13]=(__bf16)q1.y; a[14]=(__bf16)q1.z; a[15]=(__bf16)q1.w;

#pragma unroll
            for (int t = 0; t < 4; ++t)
                acc[t] = __builtin_amdgcn_wmma_f32_16x16x32_bf16(
                    /*neg_a=*/false, a, /*neg_b=*/false, bfrag[s][t],
                    /*c_mod=*/(short)0, acc[t],
                    /*reuse_a=*/false, /*reuse_b=*/false);
        }

        // C/D layout: lane<16 -> M=m0+j,    N=lane;
        //             lane>=16 -> M=m0+8+j, N=lane-16
        const int rbase = m0 + ((lane >> 4) << 3);
#pragma unroll
        for (int t = 0; t < 4; ++t) {
            const int col = t * 16 + nIdx;
#pragma unroll
            for (int j = 0; j < 8; ++j)
                hf[(size_t)(rbase + j) * GAT_OC + col] = acc[t][j];
        }
    }
}

// ------------------------------------------------------------ K2: logits
__global__ void gat_alpha(const float* __restrict__ hf,
                          const float* __restrict__ a_src,
                          const float* __restrict__ a_dst,
                          float* __restrict__ as, float* __restrict__ ad)
{
    int n = blockIdx.x * blockDim.x + threadIdx.x;
    if (n >= GAT_N) return;
    const float* hr = hf + (size_t)n * GAT_OC;
#pragma unroll
    for (int h = 0; h < 2; ++h) {
        float s = 0.f, d = 0.f;
#pragma unroll
        for (int c = 0; c < 32; ++c) {
            float v = hr[h * 32 + c];
            s += v * a_src[h * 32 + c];
            d += v * a_dst[h * 32 + c];
        }
        as[n * 2 + h] = s;
        ad[n * 2 + h] = d;
    }
}

// -------------------------------------------------------------- K3: init
__global__ void gat_init(const float* __restrict__ bias,
                         float* __restrict__ out,
                         float* __restrict__ mx, float* __restrict__ dn)
{
    int idx = blockIdx.x * blockDim.x + threadIdx.x;
    if (idx < GAT_N * GAT_OC) out[idx] = bias[idx & 63];
    if (idx < GAT_N * 2) { mx[idx] = -3.402823466e38f; dn[idx] = 0.f; }
}

__device__ __forceinline__ void edge_ids(const int* __restrict__ ei, int e,
                                         int& s, int& d)
{
    if (e < GAT_E) { s = ei[e]; d = ei[GAT_E + e]; }
    else           { s = d = e - GAT_E; }          // self-loops
}

__device__ __forceinline__ float lrelu(float v)
{
    return v > 0.f ? v : v * NEG_SLOPE;
}

// ---------------------------------------------------------- K4: seg-max
__global__ void gat_segmax(const int* __restrict__ ei,
                           const float* __restrict__ as,
                           const float* __restrict__ ad,
                           float* __restrict__ mx)
{
    int e = blockIdx.x * blockDim.x + threadIdx.x;
    if (e >= GAT_EP) return;
    int s, d; edge_ids(ei, e, s, d);
#pragma unroll
    for (int h = 0; h < 2; ++h) {
        float v = lrelu(as[s * 2 + h] + ad[d * 2 + h]);
        atomicMax(&mx[d * 2 + h], v);   // global_atomic_max_num_f32
    }
}

// ----------------------------------------------------- K5: exp + seg-sum
__global__ void gat_expsum(const int* __restrict__ ei,
                           const float* __restrict__ as,
                           const float* __restrict__ ad,
                           const float* __restrict__ mx,
                           float* __restrict__ dn,
                           float* __restrict__ ee)
{
    int e = blockIdx.x * blockDim.x + threadIdx.x;
    if (e >= GAT_EP) return;
    int s, d; edge_ids(ei, e, s, d);
#pragma unroll
    for (int h = 0; h < 2; ++h) {
        float v  = lrelu(as[s * 2 + h] + ad[d * 2 + h]);
        float ex = __expf(v - mx[d * 2 + h]);
        ee[(size_t)e * 2 + h] = ex;
        atomicAdd(&dn[d * 2 + h], ex);  // global_atomic_add_f32
    }
}

// -------------------------------------------- K6: normalize + scatter-add
__global__ void gat_scatter(const int* __restrict__ ei,
                            const float* __restrict__ ee,
                            const float* __restrict__ dn,
                            const float* __restrict__ hf,
                            float* __restrict__ out)
{
    long long idx = (long long)blockIdx.x * blockDim.x + threadIdx.x;
    if (idx >= (long long)GAT_EP * GAT_OC) return;
    int e = (int)(idx >> 6);
    int c = (int)(idx & 63);
    int h = c >> 5;
    int s, d; edge_ids(ei, e, s, d);
    float alpha = ee[(size_t)e * 2 + h] / (dn[d * 2 + h] + 1e-16f);
    atomicAdd(&out[(size_t)d * GAT_OC + c],
              alpha * hf[(size_t)s * GAT_OC + c]);
}

// ------------------------------------------------------------- launcher
extern "C" void kernel_launch(void* const* d_in, const int* in_sizes, int n_in,
                              void* d_out, int out_size, void* d_ws, size_t ws_size,
                              hipStream_t stream)
{
    const float* x     = (const float*)d_in[0];
    const int*   ei    = (const int*)  d_in[1];   // [2,E]
    const float* W     = (const float*)d_in[2];   // [128,64]
    const float* a_src = (const float*)d_in[3];   // [2,32]
    const float* a_dst = (const float*)d_in[4];   // [2,32]
    const float* bias  = (const float*)d_in[5];   // [64]
    float* out = (float*)d_out;                   // [N,64]

    float* hf = (float*)d_ws;                     // [N,64]
    float* as = hf + (size_t)GAT_N * GAT_OC;      // [N,2]
    float* ad = as + (size_t)GAT_N * 2;           // [N,2]
    float* mx = ad + (size_t)GAT_N * 2;           // [N,2]
    float* dn = mx + (size_t)GAT_N * 2;           // [N,2]
    float* ee = dn + (size_t)GAT_N * 2;           // [E+N,2]

    gat_gemm_wmma<<<256, 128, 0, stream>>>(x, W, hf);

    gat_alpha<<<(GAT_N + 255) / 256, 256, 0, stream>>>(hf, a_src, a_dst, as, ad);

    gat_init<<<(GAT_N * GAT_OC + 255) / 256, 256, 0, stream>>>(bias, out, mx, dn);

    gat_segmax<<<(GAT_EP + 255) / 256, 256, 0, stream>>>(ei, as, ad, mx);

    gat_expsum<<<(GAT_EP + 255) / 256, 256, 0, stream>>>(ei, as, ad, mx, dn, ee);

    long long total = (long long)GAT_EP * GAT_OC;
    int blocks = (int)((total + 255) / 256);
    gat_scatter<<<blocks, 256, 0, stream>>>(ei, ee, dn, hf, out);
}
